// KNN3DUNetDecoder_89756226552194
// MI455X (gfx1250) — compile-verified, matrix operand order checked by reading, not run on hardware
//
// CDNA5 / gfx1250 implementation of KNN3DUNetDecoder reference.
// GEMMs: v_wmma_f32_16x16x32_f16 (wave32), f16 activations, f32 accum, fused
// BN+LeakyReLU epilogues. B-tiles ping-pong staged into LDS with gfx1250
// async global->LDS copies (ASYNCcnt, in-order completion) so the next tile's
// copy fully overlaps the current tile's WMMAs. Gathers run against an
// L2-resident channel-contiguous f16 copy of the voxel grid.

#include <hip/hip_runtime.h>

typedef __attribute__((ext_vector_type(16))) _Float16 v16h;
typedef __attribute__((ext_vector_type(8)))  float    v8f;

union HVec { v16h v; uint4 q[2]; };
union H4   { uint2 u; _Float16 h[4]; };

constexpr int       BB   = 2;
constexpr int       NN   = 4096;
constexpr int       MM   = 4096;
constexpr int       KNk  = 16;
constexpr int       CC   = 128;
constexpr int       RR   = 32;
constexpr int       VV   = RR * RR * RR;
constexpr long long NKp  = (long long)NN * KNk;   // 65536 knn points per batch
constexpr float     SLOPE = 0.05f;
constexpr float     EPSb  = 1e-5f;

// ---------------------------------------------------------------- utilities

__global__ void zero_f_kernel(float* __restrict__ p, int n) {
  int i = blockIdx.x * blockDim.x + threadIdx.x;
  if (i < n) p[i] = 0.f;
}

__global__ void w2h_kernel(const float* __restrict__ src, _Float16* __restrict__ dst, int n) {
  int i = blockIdx.x * blockDim.x + threadIdx.x;
  if (i < n) dst[i] = (_Float16)src[i];
}

// [B,C,P] f32  ->  [B,P,C] f16   (used for feat_pt and the voxel grid)
__global__ void transpose_cp_pc_kernel(const float* __restrict__ src, _Float16* __restrict__ dst,
                                       int C, int P, long long total) {
  long long i = (long long)blockIdx.x * blockDim.x + threadIdx.x;
  if (i >= total) return;
  int pidx = (int)(i % P);
  int c    = (int)((i / P) % C);
  int b    = (int)(i / ((long long)P * C));
  dst[((long long)b * P + pidx) * C + c] = (_Float16)src[i];
}

// mean over points:  pts [B,P,3] -> sums[B*3] += pts/P
__global__ void mean_accum_kernel(const float* __restrict__ pts, float* __restrict__ sums, int P) {
  int b = blockIdx.y;
  int i = blockIdx.x * blockDim.x + threadIdx.x;
  if (i >= P) return;
  const float* q = pts + ((long long)b * P + i) * 3;
  float inv = 1.f / (float)P;
  atomicAdd(&sums[b * 3 + 0], q[0] * inv);
  atomicAdd(&sums[b * 3 + 1], q[1] * inv);
  atomicAdd(&sums[b * 3 + 2], q[2] * inv);
}

// mean of gathered knn points: inputs[B,M,3] gathered by idx[B,NK]
__global__ void mean_knn_kernel(const float* __restrict__ inputs, const int* __restrict__ idx,
                                float* __restrict__ sums) {
  int b = blockIdx.y;
  long long i = (long long)blockIdx.x * blockDim.x + threadIdx.x;
  if (i >= NKp) return;
  int j = idx[(long long)b * NKp + i];
  const float* q = inputs + ((long long)b * MM + j) * 3;
  float inv = 1.f / (float)NKp;
  atomicAdd(&sums[b * 3 + 0], q[0] * inv);
  atomicAdd(&sums[b * 3 + 1], q[1] * inv);
  atomicAdd(&sums[b * 3 + 2], q[2] * inv);
}

// --------------------------------------------------------------- KNN top-16

__global__ __launch_bounds__(256)
void knn_kernel(const float* __restrict__ p, const float* __restrict__ inputs,
                int* __restrict__ idxOut) {
  int b = blockIdx.y;
  int n = blockIdx.x * 256 + threadIdx.x;
  const float* q = p + ((long long)b * NN + n) * 3;
  float qx = q[0], qy = q[1], qz = q[2];

  float bestd[KNk];
  int   besti[KNk];
#pragma unroll
  for (int i = 0; i < KNk; ++i) { bestd[i] = 3.4e38f; besti[i] = 0; }

  __shared__ float tile[256 * 3];
  for (int j0 = 0; j0 < MM; j0 += 256) {
    int t = threadIdx.x;
#pragma unroll
    for (int r = 0; r < 3; ++r)
      tile[r * 256 + t] = inputs[((long long)b * MM + j0) * 3 + r * 256 + t];
    __syncthreads();
    for (int j = 0; j < 256; ++j) {
      float dx = qx - tile[j * 3 + 0];
      float dy = qy - tile[j * 3 + 1];
      float dz = qz - tile[j * 3 + 2];
      float d2 = dx * dx + dy * dy + dz * dz;
      if (d2 < bestd[KNk - 1]) {
        bestd[KNk - 1] = d2;
        besti[KNk - 1] = j0 + j;
#pragma unroll
        for (int i = KNk - 1; i > 0; --i) {
          if (bestd[i] < bestd[i - 1]) {
            float td = bestd[i]; bestd[i] = bestd[i - 1]; bestd[i - 1] = td;
            int   ti = besti[i]; besti[i] = besti[i - 1]; besti[i - 1] = ti;
          }
        }
      }
    }
    __syncthreads();
  }
#pragma unroll
  for (int k = 0; k < KNk; ++k)
    idxOut[(((long long)b * NN + n) * KNk) + k] = besti[k];
}

// ------------------------------------------------------------- devoxelize

__device__ __forceinline__ float vox_coord(float t, float mean) {
  float nc = ((t - mean) + 1.f) * 0.5f * (float)(RR - 1);
  return fminf(fmaxf(nc, 0.f), (float)(RR - 1));
}

// trilinear gather of 4 channels (cbase..cbase+3) from channel-contiguous grid
__device__ __forceinline__ void devox_accum(const _Float16* __restrict__ gT,
                                            float x, float y, float z,
                                            int cbase, float acc[4]) {
  float x0f = floorf(x), y0f = floorf(y), z0f = floorf(z);
  float fx = x - x0f, fy = y - y0f, fz = z - z0f;
  int x0 = (int)x0f, y0 = (int)y0f, z0 = (int)z0f;
#pragma unroll
  for (int dx = 0; dx < 2; ++dx)
#pragma unroll
    for (int dy = 0; dy < 2; ++dy)
#pragma unroll
      for (int dz = 0; dz < 2; ++dz) {
        int xi = min(x0 + dx, RR - 1);
        int yi = min(y0 + dy, RR - 1);
        int zi = min(z0 + dz, RR - 1);
        float w = (dx ? fx : 1.f - fx) * (dy ? fy : 1.f - fy) * (dz ? fz : 1.f - fz);
        const _Float16* src = gT + (long long)((xi * RR + yi) * RR + zi) * CC + cbase;
        H4 h4v; h4v.u = *(const uint2*)src;
        acc[0] += w * (float)h4v.h[0];
        acc[1] += w * (float)h4v.h[1];
        acc[2] += w * (float)h4v.h[2];
        acc[3] += w * (float)h4v.h[3];
      }
}

// one wave per point; lane covers 4 channels -> Xout[B,P,outStride] f16
__global__ __launch_bounds__(256)
void devox_points_kernel(const float* __restrict__ coords, const float* __restrict__ means,
                         const _Float16* __restrict__ gridT, _Float16* __restrict__ Xout,
                         int P, int outStride, int outOffset) {
  int b = blockIdx.y;
  int wave = threadIdx.x >> 5, lane = threadIdx.x & 31;
  int pt = blockIdx.x * 8 + wave;
  if (pt >= P) return;
  const float* q = coords + ((long long)b * P + pt) * 3;
  float cx = vox_coord(q[0], means[b * 3 + 0]);
  float cy = vox_coord(q[1], means[b * 3 + 1]);
  float cz = vox_coord(q[2], means[b * 3 + 2]);
  int cbase = lane * 4;
  float acc[4] = {0.f, 0.f, 0.f, 0.f};
  devox_accum(gridT + (long long)b * VV * CC, cx, cy, cz, cbase, acc);
  _Float16* dst = Xout + ((long long)b * P + pt) * outStride + outOffset + cbase;
  H4 o; o.h[0] = (_Float16)acc[0]; o.h[1] = (_Float16)acc[1];
  o.h[2] = (_Float16)acc[2];       o.h[3] = (_Float16)acc[3];
  *(uint2*)dst = o.u;
}

// -------------------------------------------- assemble h_in = [pfeat | vfeat_knn | pdiff]

__global__ __launch_bounds__(256)
void build_h_kernel(const float* __restrict__ p, const float* __restrict__ inputs,
                    const int* __restrict__ idx, const float* __restrict__ meansK,
                    const _Float16* __restrict__ gridT, const _Float16* __restrict__ fpt_h,
                    const float* __restrict__ pdW,
                    const float* __restrict__ pg, const float* __restrict__ pb,
                    const float* __restrict__ pm, const float* __restrict__ pv,
                    _Float16* __restrict__ h_in) {
  int b = blockIdx.y;
  int wave = threadIdx.x >> 5, lane = threadIdx.x & 31;
  long long i = (long long)blockIdx.x * 8 + wave;           // knn point index
  if (i >= NKp) return;
  int n = (int)(i / KNk);
  int j = idx[(long long)b * NKp + i];
  const float* q  = inputs + ((long long)b * MM + j) * 3;
  const float* pp = p + ((long long)b * NN + n) * 3;
  int cbase = lane * 4;
  long long orow = ((long long)b * NKp + i) * (3 * CC);

  // part 0: gathered per-point features (fpt)
  {
    const uint2* src = (const uint2*)(fpt_h + ((long long)b * MM + j) * CC + cbase);
    *(uint2*)(h_in + orow + cbase) = *src;
  }
  // part 1: devoxelized grid features at knn point
  {
    float cx = vox_coord(q[0], meansK[b * 3 + 0]);
    float cy = vox_coord(q[1], meansK[b * 3 + 1]);
    float cz = vox_coord(q[2], meansK[b * 3 + 2]);
    float acc[4] = {0.f, 0.f, 0.f, 0.f};
    devox_accum(gridT + (long long)b * VV * CC, cx, cy, cz, cbase, acc);
    H4 o; o.h[0] = (_Float16)acc[0]; o.h[1] = (_Float16)acc[1];
    o.h[2] = (_Float16)acc[2];       o.h[3] = (_Float16)acc[3];
    *(uint2*)(h_in + orow + CC + cbase) = o.u;
  }
  // part 2: cba(pdiff)  (3 -> 128 matvec + BN + leaky relu)
  {
    float d0 = pp[0] - q[0], d1 = pp[1] - q[1], d2 = pp[2] - q[2];
    H4 o;
#pragma unroll
    for (int c4 = 0; c4 < 4; ++c4) {
      int c = cbase + c4;
      float v = pdW[c * 3 + 0] * d0 + pdW[c * 3 + 1] * d1 + pdW[c * 3 + 2] * d2;
      float sc = pg[c] * rsqrtf(pv[c] + EPSb);
      v = (v - pm[c]) * sc + pb[c];
      v = v > 0.f ? v : SLOPE * v;
      o.h[c4] = (_Float16)v;
    }
    *(uint2*)(h_in + orow + 2 * CC + cbase) = o.u;
  }
}

// ----------------------------------------------------- WMMA GEMM (OC = 128)
// Y[b,p,:] = lrelu(bn(W @ X[b,p,:]))   W: [128, IC] f16, X: [B,P,IC] f16
// 8 waves: wave w owns output rows [16w,16w+16); block covers 64 points.
// Ping-pong LDS staging with gfx1250 async global->LDS copies: tile i+1's
// copy streams into the other buffer while tile i's WMMAs execute. Async
// loads complete in order, so s_wait_asynccnt 1 proves the oldest tile done.

__global__ __launch_bounds__(256)
void gemm128_bn_lrelu_kernel(const _Float16* __restrict__ Wh, const _Float16* __restrict__ X,
                             const float* __restrict__ bn_g, const float* __restrict__ bn_b,
                             const float* __restrict__ bn_m, const float* __restrict__ bn_v,
                             float* __restrict__ Yf, _Float16* __restrict__ Yh,
                             int IC, int P, int outStride, int outOffset) {
  constexpr int OC = 128;
  int b    = blockIdx.y;
  int p0   = blockIdx.x * 64;
  int tid  = threadIdx.x;
  int wave = tid >> 5;
  int lane = tid & 31;
  int ocbase = wave * 16;
  int lrow = lane & 15;
  int lhi  = lane >> 4;

  __shared__ _Float16 Xs[2][64 * 40];   // ping-pong; 64 pts x 32 k, pitch 40

  v8f acc[4] = {};
  long long xbase = ((long long)b * P + p0) * IC;

  // per-thread staging slot (constant across K loop)
  int st_pt = tid >> 2;        // 0..63
  int st_ch = tid & 3;         // 4 chunks of 8 halves
  // 32-bit LDS addresses: flat LDS pointers map to LDS by low 32 bits (ISA 10.2)
  unsigned ldsaddr[2] = {
    (unsigned)(unsigned long long)(const void*)&Xs[0][st_pt * 40 + st_ch * 8],
    (unsigned)(unsigned long long)(const void*)&Xs[1][st_pt * 40 + st_ch * 8]
  };
  const _Float16* srcBase = X + xbase + (long long)st_pt * IC + st_ch * 8;

  int nt = IC >> 5;            // K tiles of 32
  // prologue: async-stage tiles 0 and 1
  {
    unsigned long long ga0 = (unsigned long long)(srcBase);
    asm volatile("global_load_async_to_lds_b128 %0, %1, off"
                 :: "v"(ldsaddr[0]), "v"(ga0) : "memory");
    if (nt > 1) {
      unsigned long long ga1 = (unsigned long long)(srcBase + 32);
      asm volatile("global_load_async_to_lds_b128 %0, %1, off"
                   :: "v"(ldsaddr[1]), "v"(ga1) : "memory");
    }
  }

  for (int i = 0; i < nt; ++i) {
    // oldest outstanding copy == tile i (in-order completion)
    if (i + 1 < nt) asm volatile("s_wait_asynccnt 0x1" ::: "memory");
    else            asm volatile("s_wait_asynccnt 0x0" ::: "memory");
    __syncthreads();

    int kk = i << 5;
    // A fragment straight from L2-resident f16 weights (ISA 7.12.2 layout)
    HVec A;
    {
      const _Float16* wrow = Wh + (long long)(ocbase + lrow) * IC + kk + lhi * 8;
      A.q[0] = *(const uint4*)(wrow);        // K = lhi*8 .. +7
      A.q[1] = *(const uint4*)(wrow + 16);   // K = 16 + lhi*8 .. +7
    }
    const _Float16* xb = &Xs[i & 1][0];
#pragma unroll
    for (int t = 0; t < 4; ++t) {
      HVec Bf;
      const _Float16* xrow = xb + (t * 16 + lrow) * 40 + lhi * 8;
      Bf.q[0] = *(const uint4*)(xrow);
      Bf.q[1] = *(const uint4*)(xrow + 16);
      acc[t] = __builtin_amdgcn_wmma_f32_16x16x32_f16(
          false, A.v, false, Bf.v, (short)0, acc[t], false, false);
    }
    __syncthreads();   // all waves done reading buf[i&1] before it is re-staged

    if (i + 2 < nt) {
      unsigned long long ga = (unsigned long long)(srcBase + (long long)(i + 2) * 32);
      asm volatile("global_load_async_to_lds_b128 %0, %1, off"
                   :: "v"(ldsaddr[i & 1]), "v"(ga) : "memory");
      __builtin_prefetch(srcBase + (long long)(i + 2) * 32, 0, 1);
    }
  }

  // fused BN + LeakyReLU epilogue; C/D layout: oc = r + 8*lhi, pt = lane&15
#pragma unroll
  for (int t = 0; t < 4; ++t) {
    int pt = p0 + t * 16 + lrow;
    if (pt >= P) continue;
#pragma unroll
    for (int r = 0; r < 8; ++r) {
      int oc = ocbase + r + lhi * 8;
      float val = acc[t][r];
      if (bn_g) {
        float sc = bn_g[oc] * rsqrtf(bn_v[oc] + EPSb);
        val = (val - bn_m[oc]) * sc + bn_b[oc];
        val = val > 0.f ? val : SLOPE * val;
      }
      if (Yf) Yf[((long long)b * P + pt) * OC + oc] = val;
      if (Yh) Yh[((long long)b * P + pt) * outStride + outOffset + oc] = (_Float16)val;
    }
  }
}

// ------------------------------- kernel-weight softmax + weighted pooling
// logits[k] = kw_W2 . wfeat[b, n*16+k, :] + b2 ; softmax over k ;
// vfeat[c] = sum_k w[k] * h[b, n*16+k, c]  -> g_in[b,n,0:128] (f16)

__global__ __launch_bounds__(256)
void softmax_pool_kernel(const _Float16* __restrict__ wfeat, const float* __restrict__ kw_W2,
                         const float* __restrict__ kw_b2, const float* __restrict__ h_f32,
                         _Float16* __restrict__ g_in) {
  int b = blockIdx.y;
  int wave = threadIdx.x >> 5, lane = threadIdx.x & 31;
  int n = blockIdx.x * 8 + wave;
  if (n >= NN) return;
  long long base = ((long long)b * NN + n) * KNk;
  int cbase = lane * 4;
  float w2[4] = {kw_W2[cbase], kw_W2[cbase + 1], kw_W2[cbase + 2], kw_W2[cbase + 3]};
  float logits[KNk];
#pragma unroll
  for (int k = 0; k < KNk; ++k) {
    const _Float16* wf = wfeat + (base + k) * CC + cbase;
    float s = w2[0] * (float)wf[0] + w2[1] * (float)wf[1] +
              w2[2] * (float)wf[2] + w2[3] * (float)wf[3];
#pragma unroll
    for (int off = 16; off > 0; off >>= 1) s += __shfl_xor(s, off, 32);
    logits[k] = s + kw_b2[0];
  }
  float mx = logits[0];
#pragma unroll
  for (int k = 1; k < KNk; ++k) mx = fmaxf(mx, logits[k]);
  float sum = 0.f;
#pragma unroll
  for (int k = 0; k < KNk; ++k) { logits[k] = __expf(logits[k] - mx); sum += logits[k]; }
  float inv = 1.f / sum;
  float acc[4] = {0.f, 0.f, 0.f, 0.f};
#pragma unroll
  for (int k = 0; k < KNk; ++k) {
    float w = logits[k] * inv;
    const float* h = h_f32 + (base + k) * CC + cbase;
    acc[0] += w * h[0]; acc[1] += w * h[1]; acc[2] += w * h[2]; acc[3] += w * h[3];
  }
  _Float16* dst = g_in + ((long long)b * NN + n) * (2 * CC) + cbase;
  H4 o; o.h[0] = (_Float16)acc[0]; o.h[1] = (_Float16)acc[1];
  o.h[2] = (_Float16)acc[2];       o.h[3] = (_Float16)acc[3];
  *(uint2*)dst = o.u;
}

// ------------------------------------------------- final 1-channel conv

__global__ __launch_bounds__(256)
void final_dot_kernel(const _Float16* __restrict__ g_h, const float* __restrict__ W2,
                      const float* __restrict__ b2, float* __restrict__ out) {
  int b = blockIdx.y;
  int wave = threadIdx.x >> 5, lane = threadIdx.x & 31;
  int n = blockIdx.x * 8 + wave;
  if (n >= NN) return;
  int cbase = lane * 4;
  const _Float16* g = g_h + ((long long)b * NN + n) * CC + cbase;
  float s = W2[cbase + 0] * (float)g[0] + W2[cbase + 1] * (float)g[1] +
            W2[cbase + 2] * (float)g[2] + W2[cbase + 3] * (float)g[3];
#pragma unroll
  for (int off = 16; off > 0; off >>= 1) s += __shfl_xor(s, off, 32);
  if (lane == 0) out[(long long)b * NN + n] = s + b2[0];
}

// =========================================================== host launcher

extern "C" void kernel_launch(void* const* d_in, const int* in_sizes, int n_in,
                              void* d_out, int out_size, void* d_ws, size_t ws_size,
                              hipStream_t stream) {
  (void)in_sizes; (void)n_in; (void)out_size; (void)ws_size;
  const float* Pp   = (const float*)d_in[0];
  // d_in[1] = z (unused by reference)
  const float* Inp  = (const float*)d_in[2];
  const float* Fout = (const float*)d_in[3];
  const float* Fpt  = (const float*)d_in[4];
  const float* iW1  = (const float*)d_in[5];
  const float* ib1g = (const float*)d_in[6],  *ib1b = (const float*)d_in[7],
             * ib1m = (const float*)d_in[8],  *ib1v = (const float*)d_in[9];
  const float* iW2  = (const float*)d_in[10];
  const float* ib2g = (const float*)d_in[11], *ib2b = (const float*)d_in[12],
             * ib2m = (const float*)d_in[13], *ib2v = (const float*)d_in[14];
  const float* pW   = (const float*)d_in[15];
  const float* pbg  = (const float*)d_in[16], *pbb = (const float*)d_in[17],
             * pbm  = (const float*)d_in[18], *pbv = (const float*)d_in[19];
  const float* pdW  = (const float*)d_in[20];
  const float* pdg  = (const float*)d_in[21], *pdb = (const float*)d_in[22],
             * pdm  = (const float*)d_in[23], *pdv = (const float*)d_in[24];
  const float* kW1  = (const float*)d_in[25];
  const float* kb1g = (const float*)d_in[26], *kb1b = (const float*)d_in[27],
             * kb1m = (const float*)d_in[28], *kb1v = (const float*)d_in[29];
  const float* kW2  = (const float*)d_in[30];
  const float* kb2g = (const float*)d_in[31], *kb2b = (const float*)d_in[32],
             * kb2m = (const float*)d_in[33], *kb2v = (const float*)d_in[34];
  const float* kwW1 = (const float*)d_in[35];
  const float* kwbg = (const float*)d_in[36], *kwbb = (const float*)d_in[37],
             * kwbm = (const float*)d_in[38], *kwbv = (const float*)d_in[39];
  const float* kwW2 = (const float*)d_in[40];
  const float* kwb2 = (const float*)d_in[41];
  const float* bW1  = (const float*)d_in[42];
  const float* bb1g = (const float*)d_in[43], *bb1b = (const float*)d_in[44],
             * bb1m = (const float*)d_in[45], *bb1v = (const float*)d_in[46];
  const float* bW2  = (const float*)d_in[47];
  const float* bb2  = (const float*)d_in[48];
  float* out = (float*)d_out;

  // ---- workspace carve-up
  char* ws = (char*)d_ws;
  size_t off = 0;
  auto alloc = [&](size_t bytes) -> char* {
    char* ptr = ws + off;
    off += (bytes + 255) & ~(size_t)255;
    return ptr;
  };
  _Float16* gridT   = (_Float16*)alloc((size_t)BB * VV * CC * 2);      // 16.8 MB
  _Float16* iW1h    = (_Float16*)alloc(CC * CC * 2);
  _Float16* iW2h    = (_Float16*)alloc(CC * CC * 2);
  _Float16* pWh     = (_Float16*)alloc(CC * CC * 2);
  _Float16* kW1h    = (_Float16*)alloc(CC * 3 * CC * 2);
  _Float16* kW2h    = (_Float16*)alloc(CC * CC * 2);
  _Float16* kwW1h   = (_Float16*)alloc(CC * CC * 2);
  _Float16* bW1h    = (_Float16*)alloc(CC * 2 * CC * 2);
  float*    means   = (float*)alloc(12 * 4);                            // [B,3] x2
  float*    means_p = means;
  float*    means_k = means + 6;
  int*      knnIdx  = (int*)alloc((size_t)BB * NKp * 4);
  _Float16* fptT    = (_Float16*)alloc((size_t)BB * MM * CC * 2);       // feat_pt^T f16
  _Float16* fpt_h   = (_Float16*)alloc((size_t)BB * MM * CC * 2);
  _Float16* Xv      = (_Float16*)alloc((size_t)BB * NN * CC * 2);       // devox(p)
  _Float16* i1_h    = (_Float16*)alloc((size_t)BB * NN * CC * 2);
  _Float16* g_in    = (_Float16*)alloc((size_t)BB * NN * 2 * CC * 2);   // [vfeat|vfeat_pts]
  _Float16* h_in    = (_Float16*)alloc((size_t)BB * NKp * 3 * CC * 2);  // 100 MB
  _Float16* h1_h    = (_Float16*)alloc((size_t)BB * NKp * CC * 2);
  _Float16* h_h     = (_Float16*)alloc((size_t)BB * NKp * CC * 2);
  float*    h_f     = (float*)alloc((size_t)BB * NKp * CC * 4);
  _Float16* wfeat   = (_Float16*)alloc((size_t)BB * NKp * CC * 2);
  _Float16* g_h     = (_Float16*)alloc((size_t)BB * NN * CC * 2);

  // ---- weight conversion f32 -> f16
  auto w2h = [&](const float* s, _Float16* d, int n) {
    w2h_kernel<<<(n + 255) / 256, 256, 0, stream>>>(s, d, n);
  };
  w2h(iW1, iW1h, CC * CC);      w2h(iW2, iW2h, CC * CC);
  w2h(pW, pWh, CC * CC);        w2h(kW1, kW1h, CC * 3 * CC);
  w2h(kW2, kW2h, CC * CC);      w2h(kwW1, kwW1h, CC * CC);
  w2h(bW1, bW1h, CC * 2 * CC);

  // ---- grid + feat_pt transposes ([B,C,P] -> [B,P,C] f16)
  {
    long long tot = (long long)BB * CC * VV;
    transpose_cp_pc_kernel<<<(int)((tot + 255) / 256), 256, 0, stream>>>(Fout, gridT, CC, VV, tot);
  }
  {
    long long tot = (long long)BB * CC * MM;
    transpose_cp_pc_kernel<<<(int)((tot + 255) / 256), 256, 0, stream>>>(Fpt, fptT, CC, MM, tot);
  }

  // ---- means for voxel coords
  zero_f_kernel<<<1, 32, 0, stream>>>(means, 12);
  mean_accum_kernel<<<dim3(NN / 256, BB), 256, 0, stream>>>(Pp, means_p, NN);

  // ---- KNN (top-16 over 4096 db points, LDS-tiled)
  knn_kernel<<<dim3(NN / 256, BB), 256, 0, stream>>>(Pp, Inp, knnIdx);
  mean_knn_kernel<<<dim3((int)(NKp / 256), BB), 256, 0, stream>>>(Inp, knnIdx, means_k);

  // ---- fpt = cba(feat_pt, point_W)   [WMMA]
  gemm128_bn_lrelu_kernel<<<dim3(MM / 64, BB), 256, 0, stream>>>(
      pWh, fptT, pbg, pbb, pbm, pbv, nullptr, fpt_h, CC, MM, CC, 0);

  // ---- vfeat_pts path: devox(p) -> interp1 -> interp2 (into g_in cols 128..255)
  devox_points_kernel<<<dim3(NN / 8, BB), 256, 0, stream>>>(Pp, means_p, gridT, Xv, NN, CC, 0);
  gemm128_bn_lrelu_kernel<<<dim3(NN / 64, BB), 256, 0, stream>>>(
      iW1h, Xv, ib1g, ib1b, ib1m, ib1v, nullptr, i1_h, CC, NN, CC, 0);
  gemm128_bn_lrelu_kernel<<<dim3(NN / 64, BB), 256, 0, stream>>>(
      iW2h, i1_h, ib2g, ib2b, ib2m, ib2v, nullptr, g_in, CC, NN, 2 * CC, CC);

  // ---- assemble h_in = [pfeat_knn | vfeat_knn | cba(pdiff)]  (384 ch per knn point)
  build_h_kernel<<<dim3((int)(NKp / 8), BB), 256, 0, stream>>>(
      Pp, Inp, knnIdx, means_k, gridT, fpt_h, pdW, pdg, pdb, pdm, pdv, h_in);

  // ---- h = cba(cba(h_in, knn_W1), knn_W2)   [WMMA, IC=384 then 128]
  gemm128_bn_lrelu_kernel<<<dim3((int)(NKp / 64), BB), 256, 0, stream>>>(
      kW1h, h_in, kb1g, kb1b, kb1m, kb1v, nullptr, h1_h, 3 * CC, (int)NKp, CC, 0);
  gemm128_bn_lrelu_kernel<<<dim3((int)(NKp / 64), BB), 256, 0, stream>>>(
      kW2h, h1_h, kb2g, kb2b, kb2m, kb2v, h_f, h_h, CC, (int)NKp, CC, 0);

  // ---- kernel-weight branch: cba(h, kw_W1) then softmax pooling
  gemm128_bn_lrelu_kernel<<<dim3((int)(NKp / 64), BB), 256, 0, stream>>>(
      kwW1h, h_h, kwbg, kwbb, kwbm, kwbv, nullptr, wfeat, CC, (int)NKp, CC, 0);
  softmax_pool_kernel<<<dim3(NN / 8, BB), 256, 0, stream>>>(wfeat, kwW2, kwb2, h_f, g_in);

  // ---- g = cba([vfeat|vfeat_pts], blk_W1) ; net = blk_W2 . g + b2
  gemm128_bn_lrelu_kernel<<<dim3(NN / 64, BB), 256, 0, stream>>>(
      bW1h, g_in, bb1g, bb1b, bb1m, bb1v, nullptr, g_h, 2 * CC, NN, CC, 0);
  final_dot_kernel<<<dim3(NN / 8, BB), 256, 0, stream>>>(g_h, bW2, bb2, out);
}
